// Seq2SeqModel_30064771072427
// MI455X (gfx1250) — matrix operand order repaired and checked by reading, not compiled, not run on hardware
//
#include <hip/hip_runtime.h>
#include <hip/hip_bf16.h>

// ---------------------------------------------------------------------------
// Dimensions (fixed by the reference)
// ---------------------------------------------------------------------------
#define BB 64
#define SS 256
#define TT 64
#define EE 256
#define HH 256
#define VT_ 32000

typedef __attribute__((ext_vector_type(16))) _Float16 v16h;
typedef __attribute__((ext_vector_type(8)))  float    v8f;
typedef __attribute__((ext_vector_type(4)))  unsigned uv4;   // builtin 16B vector

// LDS (address_space(3)) qualified types: builtin vectors load cleanly across AS
typedef _Float16 __attribute__((address_space(3))) lds_f16_t;
typedef uv4      __attribute__((address_space(3))) lds_uv4_t;

// ---------------------------------------------------------------------------
// WMMA fragment loaders (ISA 7.12.2 layouts, wave32)
// A tile 16x32 f16: lanes 0-15 row M=l, halves[0..7]=K k0..k0+7, [8..15]=k0+16..23
//                   lanes 16-31 row M=l-16, K k0+8..15 and k0+24..31
// ---------------------------------------------------------------------------
__device__ __forceinline__ v16h load_a16(const _Float16* A, int lda, int m0, int k0) {
  int l = threadIdx.x & 31;
  const _Float16* p = A + (size_t)(m0 + (l & 15)) * lda + k0 + ((l >> 4) << 3);
  union { v16h v; uv4 u[2]; } r;
  r.u[0] = *(const uv4*)(p);
  r.u[1] = *(const uv4*)(p + 16);
  return r.v;
}

// Same A-tile loader but from an explicit LDS (AS3) pointer -> ds_load_b128
__device__ __forceinline__ v16h load_a16_lds(const lds_f16_t* A, int lda, int m0, int k0) {
  int l = threadIdx.x & 31;
  const lds_f16_t* p = A + (m0 + (l & 15)) * lda + k0 + ((l >> 4) << 3);
  union { v16h v; uv4 u[2]; } r;
  r.u[0] = *(const lds_uv4_t*)(p);
  r.u[1] = *(const lds_uv4_t*)(p + 16);
  return r.v;
}

// B tile 32x16 f16 from W stored (N x K) row-major:
// lanes 0-15: col N=n0+l, K k0..k0+7,k0+16..23 ; lanes 16-31: col n0+l-16, K k0+8..15,k0+24..31
__device__ __forceinline__ v16h load_b16(const _Float16* W, int ldw, int n0, int k0) {
  int l = threadIdx.x & 31;
  const _Float16* p = W + (size_t)(n0 + (l & 15)) * ldw + k0 + ((l >> 4) << 3);
  union { v16h v; uv4 u[2]; } r;
  r.u[0] = *(const uv4*)(p);
  r.u[1] = *(const uv4*)(p + 16);
  return r.v;
}

__device__ __forceinline__ float sigmoidf_(float x) { return 1.0f / (1.0f + __expf(-x)); }

// ---------------------------------------------------------------------------
// Generic WMMA GEMM: C(MxN,f32 and/or f16) = A(MxK,f16) * W(NxK,f16)^T + bias
// grid.x = M/16 ; grid.y = ceil(N/512) ; 256 threads = 8 waves, wave = 16x64 strip
// The 16xK A panel (shared by all 8 waves) is staged once into LDS through the
// gfx1250 async global->LDS path (ASYNCcnt), then WMMA A fragments come from
// LDS (ds_load_b128, DScnt) while B streams from global (LOADcnt).
// ---------------------------------------------------------------------------
__global__ void __launch_bounds__(256) gemm_wmma_f16(
    const _Float16* __restrict__ A, int lda,
    const _Float16* __restrict__ W, int ldw,
    const float* __restrict__ bias,
    float* __restrict__ C, _Float16* __restrict__ C16, int ldc,
    int N, int K)
{
  __shared__ __align__(16) _Float16 Ash[16 * 1024];   // 32 KB, K <= 1024

  int wave = threadIdx.x >> 5;
  int l    = threadIdx.x & 31;
  int m0   = blockIdx.x << 4;
  int n0   = (blockIdx.y << 9) + (wave << 6);

  // --- cooperative async stage of the A panel (16 rows x K halves) ---------
  {
    int rowc   = K >> 3;          // 16-byte chunks per row
    int chunks = rowc << 4;       // 16 rows
    for (int c = threadIdx.x; c < chunks; c += 256) {
      int row = c / rowc;
      int co  = c - row * rowc;
      const _Float16* gp = A + (size_t)(m0 + row) * lda + (co << 3);
      unsigned ldsoff = (unsigned)(((row * K) + (co << 3)) * 2);
      asm volatile("global_load_async_to_lds_b128 %0, %1, off"
                   :: "v"(ldsoff), "v"(gp) : "memory");
    }
    asm volatile("s_wait_asynccnt 0" ::: "memory");
    __syncthreads();
  }

  if (n0 >= N) return;            // after the barrier: wave-uniform exit

  // Launder a 32-bit LDS offset (Ash is the only LDS object -> base offset 0),
  // then rebuild an address_space(3) pointer from it. This hides the fact that
  // the panel has no compiler-visible stores (it was written by the async DMA)
  // while still lowering the fragment loads to ds_load_b128.
  unsigned ash_off = 0;
  asm volatile("" : "+v"(ash_off) :: "memory");
#pragma clang diagnostic push
#pragma clang diagnostic ignored "-Wint-to-pointer-cast"
  const lds_f16_t* ash = (const lds_f16_t*)ash_off;
#pragma clang diagnostic pop

  int half  = l >> 4;
  int col_l = l & 15;

  v8f acc[4];
#pragma unroll
  for (int nt = 0; nt < 4; ++nt) {
    int nn = n0 + nt * 16; if (nn >= N) nn = N - 16;
    float bv = bias ? bias[nn + col_l] : 0.0f;
#pragma unroll
    for (int r = 0; r < 8; ++r) acc[nt][r] = bv;
  }

  for (int k0 = 0; k0 < K; k0 += 32) {
    if (k0 + 32 < K)
      __builtin_prefetch(W + (size_t)(n0 + col_l) * ldw + k0 + 32, 0, 3);
    v16h a = load_a16_lds(ash, K, 0, k0);      // ds_load_b128 x2
    v16h bf[4];
#pragma unroll
    for (int nt = 0; nt < 4; ++nt) {
      int nn = n0 + nt * 16; if (nn >= N) nn = N - 16;
      bf[nt] = load_b16(W, ldw, nn, k0);
    }
#pragma unroll
    for (int nt = 0; nt < 4; ++nt)
      acc[nt] = __builtin_amdgcn_wmma_f32_16x16x32_f16(
          false, a, false, bf[nt], (short)0, acc[nt], false, false);
  }

#pragma unroll
  for (int nt = 0; nt < 4; ++nt) {
    int nn = n0 + nt * 16; if (nn >= N) continue;
#pragma unroll
    for (int r = 0; r < 8; ++r) {
      size_t off = (size_t)(m0 + half * 8 + r) * ldc + nn + col_l;
      if (C)   C[off]   = acc[nt][r];
      if (C16) C16[off] = (_Float16)acc[nt][r];
    }
  }
}

// ---------------------------------------------------------------------------
// Persistent LSTM direction kernel. One workgroup, 32 waves, 256 steps.
// h lives in LDS (f16, 64x256), c lives in VGPRs. Input gates G precomputed.
// ---------------------------------------------------------------------------
__global__ void __launch_bounds__(1024) lstm_encoder_dir(
    const _Float16* __restrict__ G,     // (S*B, 4H) time-major
    const _Float16* __restrict__ Whh,   // (4H, H) f16
    float* __restrict__ encOut,         // (B, S, 2H) f32
    _Float16* __restrict__ encOut16,    // (B, S, 2H) f16
    float* __restrict__ hfin, float* __restrict__ cfin,   // (B,H)
    int reverse)
{
  __shared__ __align__(16) _Float16 hbuf[BB * HH];
  for (int i = threadIdx.x; i < BB * HH; i += 1024) hbuf[i] = (_Float16)0.0f;
  __syncthreads();

  int wave  = threadIdx.x >> 5;
  int l     = threadIdx.x & 31;
  int jtile = wave >> 1;            // 0..15 (H tile)
  int mbase = (wave & 1) * 2;       // m-tiles {0,1} or {2,3}
  int half  = l >> 4;
  int col_l = l & 15;
  int colOff = reverse ? HH : 0;

  float c[2][8];
#pragma unroll
  for (int p = 0; p < 2; ++p)
#pragma unroll
    for (int r = 0; r < 8; ++r) c[p][r] = 0.0f;

  for (int t = 0; t < SS; ++t) {
    int step = reverse ? (SS - 1 - t) : t;
    const _Float16* Gt = G + (size_t)step * BB * (4 * HH);

    float hnv[2][8];

    for (int p = 0; p < 2; ++p) {
      int m0 = (mbase + p) * 16;
      v8f gate[4];
#pragma unroll
      for (int g = 0; g < 4; ++g) {
        int n0 = g * HH + jtile * 16;
#pragma unroll
        for (int r = 0; r < 8; ++r)
          gate[g][r] = (float)Gt[(size_t)(m0 + half * 8 + r) * (4 * HH) + n0 + col_l];
        for (int k0 = 0; k0 < HH; k0 += 32) {
          v16h am = load_a16(hbuf, HH, m0, k0);
          v16h bm = load_b16(Whh, HH, n0, k0);
          gate[g] = __builtin_amdgcn_wmma_f32_16x16x32_f16(
              false, am, false, bm, (short)0, gate[g], false, false);
        }
      }
#pragma unroll
      for (int r = 0; r < 8; ++r) {
        int b = m0 + half * 8 + r;
        int j = jtile * 16 + col_l;
        float iv = sigmoidf_(gate[0][r]);
        float fv = sigmoidf_(gate[1][r]);
        float gv = tanhf(gate[2][r]);
        float ov = sigmoidf_(gate[3][r]);
        float cn = fv * c[p][r] + iv * gv;
        c[p][r]  = cn;
        float hn = ov * tanhf(cn);
        hnv[p][r] = hn;
        size_t eo = ((size_t)b * SS + step) * (2 * HH) + colOff + j;
        encOut[eo]   = hn;
        encOut16[eo] = (_Float16)hn;
        if (t == SS - 1) { hfin[b * HH + j] = hn; cfin[b * HH + j] = cn; }
      }
    }

    __syncthreads();   // all waves done reading old h
#pragma unroll
    for (int p = 0; p < 2; ++p) {
      int m0 = (mbase + p) * 16;
#pragma unroll
      for (int r = 0; r < 8; ++r) {
        int b = m0 + half * 8 + r;
        hbuf[b * HH + jtile * 16 + col_l] = (_Float16)hnv[p][r];
      }
    }
    __syncthreads();   // new h visible to all
  }
}

// ---------------------------------------------------------------------------
// Attention score: score[b,s] = sum_h v[h]*tanh(enc_proj[b,s,h] + q[b,h])
// ---------------------------------------------------------------------------
__global__ void __launch_bounds__(256) attn_score_k(
    const float* __restrict__ encProj, const float* __restrict__ q,
    const float* __restrict__ v, float* __restrict__ score)
{
  int row = blockIdx.x * 8 + (threadIdx.x >> 5);
  int l   = threadIdx.x & 31;
  int b   = row >> 8;                 // S = 256
  const float* ep = encProj + (size_t)row * HH;
  const float* qb = q + b * HH;
  float s = 0.0f;
#pragma unroll
  for (int i = 0; i < 8; ++i) {
    int h = l + i * 32;
    s += v[h] * tanhf(ep[h] + qb[h]);
  }
#pragma unroll
  for (int o = 16; o > 0; o >>= 1) s += __shfl_xor(s, o, 32);
  if (l == 0) score[row] = s;
}

// softmax over S per batch row, then ctx = a @ enc_out -> f16 into Xcat[:,256:768]
__global__ void __launch_bounds__(256) softmax_ctx_k(
    const float* __restrict__ score, const float* __restrict__ encOut,
    _Float16* __restrict__ xcat)
{
  __shared__ float red[256];
  __shared__ float al[256];
  int b = blockIdx.x, tid = threadIdx.x;
  float sc = score[b * SS + tid];
  red[tid] = sc; __syncthreads();
  for (int o = 128; o > 0; o >>= 1) { if (tid < o) red[tid] = fmaxf(red[tid], red[tid + o]); __syncthreads(); }
  float mx = red[0]; __syncthreads();
  float e = __expf(sc - mx);
  red[tid] = e; __syncthreads();
  for (int o = 128; o > 0; o >>= 1) { if (tid < o) red[tid] += red[tid + o]; __syncthreads(); }
  float inv = 1.0f / red[0];
  al[tid] = e * inv; __syncthreads();

  float a0 = 0.0f, a1 = 0.0f;
  const float* eb = encOut + (size_t)b * SS * (2 * HH);
  for (int s = 0; s < SS; ++s) {
    float a = al[s];
    a0 += a * eb[s * 512 + tid];
    a1 += a * eb[s * 512 + 256 + tid];
  }
  xcat[b * 1024 + 256 + tid] = (_Float16)a0;
  xcat[b * 1024 + 512 + tid] = (_Float16)a1;
}

// Xcat[b, 0:256] = tgt embedding at step t (time-major)
__global__ void copy_emb_k(const _Float16* __restrict__ Xtgt, int t,
                           _Float16* __restrict__ xcat)
{
  int b = blockIdx.x, j = threadIdx.x;
  xcat[b * 1024 + j] = Xtgt[((size_t)t * BB + b) * EE + j];
}

// LSTM cell elementwise: updates c, scatters new h (f16) to up to two slots
__global__ void lstm_cell_k(const float* __restrict__ gates, float* __restrict__ c,
                            _Float16* __restrict__ hd0, int ld0,
                            _Float16* __restrict__ hd1, int ld1)
{
  int b = blockIdx.x, j = threadIdx.x;
  const float* g = gates + b * 1024;
  float iv = sigmoidf_(g[j]);
  float fv = sigmoidf_(g[j + 256]);
  float gv = tanhf(g[j + 512]);
  float ov = sigmoidf_(g[j + 768]);
  float cn = fv * c[b * HH + j] + iv * gv;
  c[b * HH + j] = cn;
  _Float16 hn = (_Float16)(ov * tanhf(cn));
  if (hd0) hd0[b * ld0 + j] = hn;
  if (hd1) hd1[b * ld1 + j] = hn;
}

// decoder init: h=hf+hb, c=cf+cb
__global__ void init_dec_k(const float* hf, const float* hb,
                           const float* cf, const float* cb,
                           float* c0, float* c1,
                           _Float16* xcat, _Float16* A1)
{
  int b = blockIdx.x, j = threadIdx.x;
  float h = hf[b * HH + j] + hb[b * HH + j];
  float c = cf[b * HH + j] + cb[b * HH + j];
  c0[b * HH + j] = c; c1[b * HH + j] = c;
  _Float16 h16 = (_Float16)h;
  xcat[b * 1024 + 768 + j] = h16;   // h0 slot for dec0 GEMM
  A1[b * 512 + j]          = h16;   // h0 slot for dec1 GEMM
  A1[b * 512 + 256 + j]    = h16;   // h1 slot (also q-proj A)
}

// utility converters
__global__ void f32_to_f16_k(const float* __restrict__ s, _Float16* __restrict__ d, int n) {
  int i = blockIdx.x * 256 + threadIdx.x;
  if (i < n) d[i] = (_Float16)s[i];
}
__global__ void pack2_f16_k(const float* __restrict__ W1, int K1,
                            const float* __restrict__ W2, int K2,
                            _Float16* __restrict__ d, int n) {
  int i = blockIdx.x * 256 + threadIdx.x;
  if (i >= n) return;
  int Kt = K1 + K2, r = i / Kt, cc = i % Kt;
  d[i] = (_Float16)(cc < K1 ? W1[(size_t)r * K1 + cc] : W2[(size_t)r * K2 + cc - K1]);
}
// embedding gather -> time-major f16 (S,B,E)
__global__ void embed_gather_k(const int* __restrict__ idx, const float* __restrict__ emb,
                               _Float16* __restrict__ out, int Slen) {
  int b = blockIdx.x / Slen, s = blockIdx.x % Slen;
  int j = threadIdx.x;
  int tok = idx[b * Slen + s];
  out[((size_t)s * BB + b) * EE + j] = (_Float16)emb[(size_t)tok * EE + j];
}

// ---------------------------------------------------------------------------
extern "C" void kernel_launch(void* const* d_in, const int* in_sizes, int n_in,
                              void* d_out, int out_size, void* d_ws, size_t ws_size,
                              hipStream_t stream) {
  const int*   src      = (const int*)d_in[0];
  const int*   tgt      = (const int*)d_in[1];
  const float* src_emb  = (const float*)d_in[2];
  const float* tgt_emb  = (const float*)d_in[3];
  const float* fWih     = (const float*)d_in[4];
  const float* fWhh     = (const float*)d_in[5];
  const float* fb       = (const float*)d_in[6];
  const float* bWih     = (const float*)d_in[7];
  const float* bWhh     = (const float*)d_in[8];
  const float* bb       = (const float*)d_in[9];
  const float* d0Wih    = (const float*)d_in[10];
  const float* d0Whh    = (const float*)d_in[11];
  const float* d0b      = (const float*)d_in[12];
  const float* d1Wih    = (const float*)d_in[13];
  const float* d1Whh    = (const float*)d_in[14];
  const float* d1b      = (const float*)d_in[15];
  const float* attnW    = (const float*)d_in[16];
  const float* attnB    = (const float*)d_in[17];
  const float* attnV    = (const float*)d_in[18];
  const float* projW    = (const float*)d_in[19];
  const float* projB    = (const float*)d_in[20];
  float* out = (float*)d_out;

  // workspace carve-up
  char* w = (char*)d_ws; size_t off = 0;
  auto alloc = [&](size_t bytes) -> void* {
    void* p = w + off; off += bytes; off = (off + 255) & ~(size_t)255; return p;
  };
  _Float16* Xsrc16  = (_Float16*)alloc((size_t)SS * BB * EE * 2);
  _Float16* Xtgt16  = (_Float16*)alloc((size_t)TT * BB * EE * 2);
  _Float16* Wfih16  = (_Float16*)alloc((size_t)1024 * 256 * 2);
  _Float16* Wfhh16  = (_Float16*)alloc((size_t)1024 * 256 * 2);
  _Float16* Wbih16  = (_Float16*)alloc((size_t)1024 * 256 * 2);
  _Float16* Wbhh16  = (_Float16*)alloc((size_t)1024 * 256 * 2);
  _Float16* W0cat   = (_Float16*)alloc((size_t)1024 * 1024 * 2);
  _Float16* W1cat   = (_Float16*)alloc((size_t)1024 * 512 * 2);
  _Float16* Wattn16 = (_Float16*)alloc((size_t)256 * 768 * 2);
  _Float16* Wproj16 = (_Float16*)alloc((size_t)VT_ * 256 * 2);
  _Float16* Gf      = (_Float16*)alloc((size_t)SS * BB * 1024 * 2);
  _Float16* Gb      = (_Float16*)alloc((size_t)SS * BB * 1024 * 2);
  float*    EncOut  = (float*)   alloc((size_t)BB * SS * 512 * 4);
  _Float16* EncOut16= (_Float16*)alloc((size_t)BB * SS * 512 * 2);
  float*    EncProj = (float*)   alloc((size_t)BB * SS * 256 * 4);
  float*    hfinF   = (float*)alloc(BB * HH * 4);
  float*    cfinF   = (float*)alloc(BB * HH * 4);
  float*    hfinB   = (float*)alloc(BB * HH * 4);
  float*    cfinB   = (float*)alloc(BB * HH * 4);
  float*    qbuf    = (float*)alloc(BB * HH * 4);
  float*    score   = (float*)alloc(BB * SS * 4);
  _Float16* Xcat    = (_Float16*)alloc((size_t)BB * 1024 * 2);
  _Float16* A1      = (_Float16*)alloc((size_t)BB * 512 * 2);
  float*    gates0  = (float*)alloc((size_t)BB * 1024 * 4);
  float*    gates1  = (float*)alloc((size_t)BB * 1024 * 4);
  float*    c0      = (float*)alloc(BB * HH * 4);
  float*    c1      = (float*)alloc(BB * HH * 4);
  _Float16* DecOut16= (_Float16*)alloc((size_t)BB * TT * HH * 2);
  (void)ws_size; (void)in_sizes; (void)n_in; (void)out_size;

  // --- weight conversion / packing ---
  f32_to_f16_k<<<1024, 256, 0, stream>>>(fWih, Wfih16, 1024 * 256);
  f32_to_f16_k<<<1024, 256, 0, stream>>>(fWhh, Wfhh16, 1024 * 256);
  f32_to_f16_k<<<1024, 256, 0, stream>>>(bWih, Wbih16, 1024 * 256);
  f32_to_f16_k<<<1024, 256, 0, stream>>>(bWhh, Wbhh16, 1024 * 256);
  pack2_f16_k<<<4096, 256, 0, stream>>>(d0Wih, 768, d0Whh, 256, W0cat, 1024 * 1024);
  pack2_f16_k<<<2048, 256, 0, stream>>>(d1Wih, 256, d1Whh, 256, W1cat, 1024 * 512);
  f32_to_f16_k<<<768, 256, 0, stream>>>(attnW, Wattn16, 256 * 768);
  f32_to_f16_k<<<32000, 256, 0, stream>>>(projW, Wproj16, VT_ * 256);

  // --- embeddings (time-major) ---
  embed_gather_k<<<BB * SS, 256, 0, stream>>>(src, src_emb, Xsrc16, SS);
  embed_gather_k<<<BB * TT, 256, 0, stream>>>(tgt, tgt_emb, Xtgt16, TT);

  // --- hoisted input-gate GEMMs: G = x @ Wih^T + b  (M=16384,N=1024,K=256) ---
  gemm_wmma_f16<<<dim3(1024, 2), 256, 0, stream>>>(Xsrc16, 256, Wfih16, 256, fb,
                                                   nullptr, Gf, 1024, 1024, 256);
  gemm_wmma_f16<<<dim3(1024, 2), 256, 0, stream>>>(Xsrc16, 256, Wbih16, 256, bb,
                                                   nullptr, Gb, 1024, 1024, 256);

  // --- persistent recurrent encoders (1 WGP each, 256 steps) ---
  lstm_encoder_dir<<<1, 1024, 0, stream>>>(Gf, Wfhh16, EncOut, EncOut16, hfinF, cfinF, 0);
  lstm_encoder_dir<<<1, 1024, 0, stream>>>(Gb, Wbhh16, EncOut, EncOut16, hfinB, cfinB, 1);

  // --- enc_proj = enc_out @ Wa_e^T + attn_b  (M=16384,N=256,K=512) ---
  gemm_wmma_f16<<<dim3(1024, 1), 256, 0, stream>>>(EncOut16, 512, Wattn16 + 256, 768,
                                                   attnB, EncProj, nullptr, 256, 256, 512);

  init_dec_k<<<BB, 256, 0, stream>>>(hfinF, hfinB, cfinF, cfinB, c0, c1, Xcat, A1);

  // --- decoder steps ---
  for (int t = 0; t < TT; ++t) {
    // q = h1 @ Wa_h^T
    gemm_wmma_f16<<<dim3(4, 1), 256, 0, stream>>>(A1 + 256, 512, Wattn16, 768,
                                                  nullptr, qbuf, nullptr, 256, 256, 256);
    attn_score_k<<<BB * SS / 8, 256, 0, stream>>>(EncProj, qbuf, attnV, score);
    softmax_ctx_k<<<BB, 256, 0, stream>>>(score, EncOut, Xcat);
    copy_emb_k<<<BB, 256, 0, stream>>>(Xtgt16, t, Xcat);
    // gates0 = [emb|ctx|h0] @ [W0ih|W0hh]^T + b0   (M=64,N=1024,K=1024)
    gemm_wmma_f16<<<dim3(4, 2), 256, 0, stream>>>(Xcat, 1024, W0cat, 1024, d0b,
                                                  gates0, nullptr, 1024, 1024, 1024);
    lstm_cell_k<<<BB, 256, 0, stream>>>(gates0, c0, Xcat + 768, 1024, A1, 512);
    // gates1 = [h0n|h1] @ [W1ih|W1hh]^T + b1   (M=64,N=1024,K=512)
    gemm_wmma_f16<<<dim3(4, 2), 256, 0, stream>>>(A1, 512, W1cat, 512, d1b,
                                                  gates1, nullptr, 1024, 1024, 512);
    lstm_cell_k<<<BB, 256, 0, stream>>>(gates1, c1, A1 + 256, 512,
                                        DecOut16 + t * HH, TT * HH);
  }

  // --- final projection: logits = dec_out @ projW^T + b  (M=4096,N=32000,K=256) ---
  gemm_wmma_f16<<<dim3(256, 63), 256, 0, stream>>>(DecOut16, 256, Wproj16, 256, projB,
                                                   out, nullptr, VT_, VT_, 256);
}
